// HeteroSAGE_592705486889
// MI455X (gfx1250) — matrix operand order repaired and checked by reading, not compiled, not run on hardware
//
#include <hip/hip_runtime.h>
#include <stdint.h>

// ---------------------------------------------------------------------------
// Types for CDNA5 WMMA (wave32): 16x16x32 bf16 -> f32
// ---------------------------------------------------------------------------
typedef __bf16 bf16_t;
typedef bf16_t v16bf __attribute__((ext_vector_type(16)));
typedef float  v8f   __attribute__((ext_vector_type(8)));
typedef unsigned short u16;

static __device__ __forceinline__ u16 f32_to_bf16(float f) {
  union { float f; unsigned u; } x; x.f = f;
  unsigned r = x.u + 0x7FFFu + ((x.u >> 16) & 1u);   // round-to-nearest-even
  return (u16)(r >> 16);
}

static __device__ __forceinline__ v8f wmma_bf16(v16bf a, v16bf b, v8f c) {
  return __builtin_amdgcn_wmma_f32_16x16x32_bf16(
      /*neg_a=*/false, a, /*neg_b=*/false, b,
      /*c_mod=*/(short)0, c, /*reuse_a=*/false, /*reuse_b=*/false);
}

static __device__ __forceinline__ float sigmoidf_(float x) {
  return 1.0f / (1.0f + __expf(-x));
}

// ---------------------------------------------------------------------------
// Packed fragment layouts (ISA 7.12.2, wave32):
//  A (16x32 bf16 tile): lane l (m = l&15, khalf = l>>4) holds 16 bf16:
//     elem e -> K = kc*32 + ((e>=8)?16:0) + 8*khalf + (e&7)
//  Flat A storage (v16bf units): (rowTile*kchunks + kc)*32 + lane
//  B (32x16 bf16 tile): lane l (n = l&15) holds K = kc*32 + ((e>=8)?16:0) + 8*(l>>4) + (e&7)
//  Flat B storage (v16bf units): (kc*ntiles + colTile)*32 + lane
//  C/D (16x16 f32): lane l col n = l&15; elem e row m = e + 8*(l>>4)
// ---------------------------------------------------------------------------

// ---- pack a K x M fp32 weight (optionally stacked [s0;s1]) into B layout ----
__global__ void k_pack_b2(const float* __restrict__ s0, const float* __restrict__ s1,
                          int K0, int K, int M, u16* __restrict__ dst) {
  int idx = blockIdx.x * blockDim.x + threadIdx.x;
  int total = K * M;
  if (idx >= total) return;
  int ntiles = M >> 4;
  int e = idx & 15, lane = (idx >> 4) & 31;
  int grp = idx >> 9;
  int nt = grp % ntiles, kc = grp / ntiles;
  int n = nt * 16 + (lane & 15);
  int k = kc * 32 + ((e >> 3) << 4) + ((lane >> 4) << 3) + (e & 7);
  float v = (k < K0) ? s0[(size_t)k * M + n] : s1[(size_t)(k - K0) * M + n];
  dst[idx] = f32_to_bf16(v);
}

// ---- pack row-major fp32 activations (nrows x K) into A layout ----
__global__ void k_pack_a_f32(const float* __restrict__ src, u16* __restrict__ dst,
                             int nrows, int K) {
  int idx = blockIdx.x * blockDim.x + threadIdx.x;
  int total = nrows * K;
  if (idx >= total) return;
  int kchunks = K >> 5;
  int e = idx & 15, lane = (idx >> 4) & 31;
  int grp = idx >> 9;
  int kc = grp % kchunks, tile = grp / kchunks;
  int r = tile * 16 + (lane & 15);
  int k = kc * 32 + ((e >> 3) << 4) + ((lane >> 4) << 3) + (e & 7);
  dst[idx] = f32_to_bf16(src[(size_t)r * K + k]);
}

// ---- embedding concat (K=384) gathered straight into A layout ----
__global__ void k_pack_embed(const int* __restrict__ feats,
                             const float* __restrict__ E0, const float* __restrict__ E1,
                             const float* __restrict__ E2, u16* __restrict__ dst, int nrows) {
  int idx = blockIdx.x * blockDim.x + threadIdx.x;
  int total = nrows * 384;
  if (idx >= total) return;
  const int kchunks = 12;
  int e = idx & 15, lane = (idx >> 4) & 31;
  int grp = idx >> 9;
  int kc = grp % kchunks, tile = grp / kchunks;
  int r = tile * 16 + (lane & 15);
  int k = kc * 32 + ((e >> 3) << 4) + ((lane >> 4) << 3) + (e & 7);
  int which = k >> 7, kk = k & 127;
  int f = feats[(size_t)r * 3 + which];
  const float* E = (which == 0) ? E0 : ((which == 1) ? E1 : E2);
  dst[idx] = f32_to_bf16(E[(size_t)f * 128 + kk]);
}

// ---- [x | s1] concat (K=256) into A layout. nbr!=null -> x = gather(hb, nbr[:,t]) ----
__global__ void k_pack_concat(const u16* __restrict__ hb, const int* __restrict__ nbr,
                              int t, int D, const float* __restrict__ s1,
                              u16* __restrict__ dst, int nrows) {
  int idx = blockIdx.x * blockDim.x + threadIdx.x;
  int total = nrows * 256;
  if (idx >= total) return;
  int e = idx & 15, lane = (idx >> 4) & 31;
  int grp = idx >> 9;
  int kc = grp & 7, tile = grp >> 3;
  int r = tile * 16 + (lane & 15);
  int k = kc * 32 + ((e >> 3) << 4) + ((lane >> 4) << 3) + (e & 7);
  u16 v;
  if (k < 128) {
    int rr = nbr ? nbr[(size_t)r * D + t] : r;
    v = hb[(size_t)rr * 128 + k];                      // already bf16 bits
  } else {
    v = f32_to_bf16(s1[(size_t)r * 128 + (k - 128)]);
  }
  dst[idx] = v;
}

// ---- generic WMMA GEMM: C[nrows x M] (+bias)(relu?)(accum?), A/B packed bf16 ----
__global__ void k_gemm(const u16* __restrict__ A, const u16* __restrict__ B,
                       const float* __restrict__ bias, float* __restrict__ C,
                       int nrows, int K, int M, int relu, int accum) {
  int lane = threadIdx.x & 31;
  int w = threadIdx.x >> 5;
  int ntiles = M >> 4;
  int total_tiles = (nrows >> 4) * ntiles;
  int gwid = blockIdx.x * (blockDim.x >> 5) + w;
  if (gwid >= total_tiles) return;                     // wave-uniform
  int row_tile = gwid / ntiles, col_tile = gwid % ntiles;
  int kchunks = K >> 5;
  const v16bf* Ap = (const v16bf*)A + ((size_t)row_tile * kchunks) * 32 + lane;
  const v16bf* Bp = (const v16bf*)B;
  v8f acc = {};
  for (int kc = 0; kc < kchunks; ++kc) {
    v16bf a = Ap[(size_t)kc * 32];
    v16bf b = Bp[((size_t)kc * ntiles + col_tile) * 32 + lane];
    acc = wmma_bf16(a, b, acc);
  }
  int n = col_tile * 16 + (lane & 15);
  int mbase = row_tile * 16 + ((lane >> 4) << 3);
  float bv = bias ? bias[n] : 0.0f;
#pragma unroll
  for (int e = 0; e < 8; ++e) {
    float v = acc[e] + bv;
    if (relu) v = fmaxf(v, 0.0f);
    size_t o = (size_t)(mbase + e) * M + n;
    if (accum) C[o] += v; else C[o] = v;
  }
}

// ---- fused LSTM step: gates = [x|hc] @ [Wih;Whh] + (bih+bhh); cell update ----
// block = 256 threads (8 waves); wave w owns hidden cols [16w,16w+16) for all 4 gates.
__global__ void k_lstm_step(const u16* __restrict__ packA, const u16* __restrict__ packB,
                            const float* __restrict__ bih, const float* __restrict__ bhh,
                            float* __restrict__ hc, float* __restrict__ cc,
                            const unsigned char* __restrict__ mask, int t, int D) {
  int lane = threadIdx.x & 31;
  int w = threadIdx.x >> 5;                            // 0..7
  int row_tile = blockIdx.x;
  const int kchunks = 8;                               // K = 256
  const v16bf* Ap = (const v16bf*)packA + ((size_t)row_tile * kchunks) * 32 + lane;
  const v16bf* Bp = (const v16bf*)packB;
  v8f ai = {}, af = {}, ag = {}, ao = {};
  for (int kc = 0; kc < kchunks; ++kc) {
    v16bf a = Ap[(size_t)kc * 32];
    if (kc + 1 < kchunks)
      __builtin_prefetch((const void*)(Ap + (size_t)(kc + 1) * 32), 0, 0);
    const v16bf* brow = Bp + (size_t)kc * 1024 + lane; // 32 col-tiles * 32 lanes
    v16bf b0 = brow[(size_t)(w)      * 32];            // gate i
    v16bf b1 = brow[(size_t)(w + 8)  * 32];            // gate f
    v16bf b2 = brow[(size_t)(w + 16) * 32];            // gate g
    v16bf b3 = brow[(size_t)(w + 24) * 32];            // gate o
    ai = wmma_bf16(a, b0, ai);
    af = wmma_bf16(a, b1, af);
    ag = wmma_bf16(a, b2, ag);
    ao = wmma_bf16(a, b3, ao);
  }
  int c = w * 16 + (lane & 15);
  float bi_ = bih[c]       + bhh[c];
  float bf_ = bih[c + 128] + bhh[c + 128];
  float bg_ = bih[c + 256] + bhh[c + 256];
  float bo_ = bih[c + 384] + bhh[c + 384];
  int mbase = row_tile * 16 + ((lane >> 4) << 3);
#pragma unroll
  for (int e = 0; e < 8; ++e) {
    int r = mbase + e;
    size_t o = (size_t)r * 128 + c;
    float iv = sigmoidf_(ai[e] + bi_);
    float fv = sigmoidf_(af[e] + bf_);
    float gv = tanhf(ag[e] + bg_);
    float ov = sigmoidf_(ao[e] + bo_);
    float c_old = cc[o], h_old = hc[o];
    float c_new = fv * c_old + iv * gv;
    float h_new = ov * tanhf(c_new);
    bool m = mask[(size_t)r * D + t] != 0;
    hc[o] = m ? h_new : h_old;
    cc[o] = m ? c_new : c_old;
  }
}

// ---- masked mean aggregation over D neighbors ----
__global__ void k_agg(const float* __restrict__ h, const int* __restrict__ nbr,
                      const unsigned char* __restrict__ mask, float* __restrict__ agg,
                      int N, int D) {
  int idx = blockIdx.x * blockDim.x + threadIdx.x;
  if (idx >= N * 128) return;
  int r = idx >> 7, k = idx & 127;
  float s = 0.0f; int cnt = 0;
  for (int d = 0; d < D; ++d) {
    if (mask[(size_t)r * D + d]) { s += h[(size_t)nbr[(size_t)r * D + d] * 128 + k]; ++cnt; }
  }
  agg[idx] = s / (float)max(cnt, 1);
}

// ---- h = relu(LN(sum/3 + h)); writes f32 master + bf16 copy ----
__global__ void k_combine_ln(const float* __restrict__ sum3, float* __restrict__ h,
                             u16* __restrict__ hb, const float* __restrict__ g,
                             const float* __restrict__ b) {
  int r = blockIdx.x, k = threadIdx.x;                 // 128 threads
  size_t o = (size_t)r * 128 + k;
  float v = sum3[o] * (1.0f / 3.0f) + h[o];
  __shared__ float red[128];
  red[k] = v; __syncthreads();
  for (int s = 64; s > 0; s >>= 1) { if (k < s) red[k] += red[k + s]; __syncthreads(); }
  float mu = red[0] * (1.0f / 128.0f); __syncthreads();
  float d = v - mu;
  red[k] = d * d; __syncthreads();
  for (int s = 64; s > 0; s >>= 1) { if (k < s) red[k] += red[k + s]; __syncthreads(); }
  float var = red[0] * (1.0f / 128.0f);
  float y = d * rsqrtf(var + 1e-5f) * g[k] + b[k];
  y = fmaxf(y, 0.0f);
  h[o] = y;
  hb[o] = f32_to_bf16(y);
}

// ---- f32 row-major -> bf16 row-major ----
__global__ void k_cast_bf16(const float* __restrict__ src, u16* __restrict__ dst, int total) {
  int idx = blockIdx.x * blockDim.x + threadIdx.x;
  if (idx < total) dst[idx] = f32_to_bf16(src[idx]);
}

// ---- segment max (values >= 0 post-relu => uint atomicMax == float max) ----
__global__ void k_segmax(const float* __restrict__ h, const int* __restrict__ gid,
                         float* __restrict__ pooled, int N) {
  int idx = blockIdx.x * blockDim.x + threadIdx.x;
  if (idx >= N * 128) return;
  int r = idx >> 7, k = idx & 127;
  unsigned v = __float_as_uint(h[idx]);
  atomicMax((unsigned*)&pooled[(size_t)gid[r] * 128 + k], v);
}

// ---------------------------------------------------------------------------
// Host orchestration
// ---------------------------------------------------------------------------
static inline void launch_gemm(const u16* A, const u16* B, const float* bias, float* C,
                               int nrows, int K, int M, int relu, int accum, hipStream_t s) {
  int tiles = (nrows / 16) * (M / 16);
  int blocks = (tiles + 3) / 4;
  hipLaunchKernelGGL(k_gemm, dim3(blocks), dim3(128), 0, s, A, B, bias, C, nrows, K, M, relu, accum);
}
static inline int nb(int total) { return (total + 255) / 256; }

extern "C" void kernel_launch(void* const* d_in, const int* in_sizes, int n_in,
                              void* d_out, int out_size, void* d_ws, size_t ws_size,
                              hipStream_t stream) {
  const int N = 60000, D = 5, G = 2048, NL = 2;

  const int*  node_feats = (const int*)d_in[0];
  const int*  nbr_fwd  = (const int*)d_in[1];
  const unsigned char* mask_fwd = (const unsigned char*)d_in[2];
  const int*  nbr_bwd  = (const int*)d_in[3];
  const unsigned char* mask_bwd = (const unsigned char*)d_in[4];
  const int*  nbr_rep  = (const int*)d_in[5];
  const unsigned char* mask_rep = (const unsigned char*)d_in[6];
  const int*  graph_ids = (const int*)d_in[7];
  const float *E0 = (const float*)d_in[8], *E1 = (const float*)d_in[9], *E2 = (const float*)d_in[10];
  const float *W1 = (const float*)d_in[11], *b1 = (const float*)d_in[12];
  const float *W2 = (const float*)d_in[13], *b2 = (const float*)d_in[14];
  const float *W3 = (const float*)d_in[15], *b3 = (const float*)d_in[16];
  const float *Wc1 = (const float*)d_in[55], *bc1 = (const float*)d_in[56];
  const float *Wc2 = (const float*)d_in[57], *bc2 = (const float*)d_in[58];
  const float *Wc3 = (const float*)d_in[59], *bc3 = (const float*)d_in[60];

  // workspace carve-up (256B aligned)
  char* ws = (char*)d_ws;
  size_t off = 0;
  auto take = [&](size_t bytes) { size_t o = off; off = (off + bytes + 255) & ~(size_t)255; return o; };
  float* h_f32  = (float*)(ws + take((size_t)N * 128 * 4));
  u16*   h_bf16 = (u16*)  (ws + take((size_t)N * 128 * 2));
  u16*   packA  = (u16*)  (ws + take((size_t)N * 384 * 2));
  float* tmpC   = (float*)(ws + take((size_t)N * 256 * 4));   // proj stage 1 out; later agg buffer
  float* sum3   = (float*)(ws + take((size_t)N * 128 * 4));
  float* hc     = (float*)(ws + take((size_t)N * 128 * 4));
  float* cc     = (float*)(ws + take((size_t)N * 128 * 4));
  u16* pw1 = (u16*)(ws + take((size_t)384 * 256 * 2));
  u16* pw2 = (u16*)(ws + take((size_t)256 * 128 * 2));
  u16* pw3 = (u16*)(ws + take((size_t)128 * 128 * 2));
  u16* pwih[4]; for (int i = 0; i < 4; ++i) pwih[i] = (u16*)(ws + take((size_t)256 * 512 * 2));
  u16* pwsn[6]; for (int i = 0; i < 6; ++i) pwsn[i] = (u16*)(ws + take((size_t)256 * 128 * 2));
  u16* pc1 = (u16*)(ws + take((size_t)128 * 128 * 2));
  u16* pc2 = (u16*)(ws + take((size_t)128 * 128 * 2));
  u16* pc3 = (u16*)(ws + take((size_t)128 * 32 * 2));
  float* pooled = (float*)(ws + take((size_t)G * 128 * 4));
  float* z1     = (float*)(ws + take((size_t)G * 128 * 4));
  float* z2     = (float*)(ws + take((size_t)G * 128 * 4));
  u16*   packG  = (u16*)  (ws + take((size_t)G * 128 * 2));
  (void)ws_size; (void)in_sizes; (void)n_in; (void)out_size;

  // ---- pack weights (bf16 fragment layout) ----
  hipLaunchKernelGGL(k_pack_b2, dim3(nb(384*256)), dim3(256), 0, stream, W1, W1, 384, 384, 256, pw1);
  hipLaunchKernelGGL(k_pack_b2, dim3(nb(256*128)), dim3(256), 0, stream, W2, W2, 256, 256, 128, pw2);
  hipLaunchKernelGGL(k_pack_b2, dim3(nb(128*128)), dim3(256), 0, stream, W3, W3, 128, 128, 128, pw3);
  for (int L = 0; L < NL; ++L) {
    int base = 17 + L * 19;
    const float* Wih_f = (const float*)d_in[base + 0];
    const float* Whh_f = (const float*)d_in[base + 1];
    const float* Ws_f  = (const float*)d_in[base + 4];
    const float* Wn_f  = (const float*)d_in[base + 5];
    const float* Wih_b = (const float*)d_in[base + 7];
    const float* Whh_b = (const float*)d_in[base + 8];
    const float* Ws_b  = (const float*)d_in[base + 11];
    const float* Wn_b  = (const float*)d_in[base + 12];
    const float* Ws_r  = (const float*)d_in[base + 14];
    const float* Wn_r  = (const float*)d_in[base + 15];
    hipLaunchKernelGGL(k_pack_b2, dim3(nb(256*512)), dim3(256), 0, stream, Wih_f, Whh_f, 128, 256, 512, pwih[L*2+0]);
    hipLaunchKernelGGL(k_pack_b2, dim3(nb(256*512)), dim3(256), 0, stream, Wih_b, Whh_b, 128, 256, 512, pwih[L*2+1]);
    hipLaunchKernelGGL(k_pack_b2, dim3(nb(256*128)), dim3(256), 0, stream, Ws_f, Wn_f, 128, 256, 128, pwsn[L*3+0]);
    hipLaunchKernelGGL(k_pack_b2, dim3(nb(256*128)), dim3(256), 0, stream, Ws_b, Wn_b, 128, 256, 128, pwsn[L*3+1]);
    hipLaunchKernelGGL(k_pack_b2, dim3(nb(256*128)), dim3(256), 0, stream, Ws_r, Wn_r, 128, 256, 128, pwsn[L*3+2]);
  }
  hipLaunchKernelGGL(k_pack_b2, dim3(nb(128*128)), dim3(256), 0, stream, Wc1, Wc1, 128, 128, 128, pc1);
  hipLaunchKernelGGL(k_pack_b2, dim3(nb(128*128)), dim3(256), 0, stream, Wc2, Wc2, 128, 128, 128, pc2);
  hipLaunchKernelGGL(k_pack_b2, dim3(nb(128*32)),  dim3(256), 0, stream, Wc3, Wc3, 128, 128, 32,  pc3);

  // ---- embedding + projection MLP: 384 -> 256 -> 128 -> 128 ----
  hipLaunchKernelGGL(k_pack_embed, dim3(nb(N*384)), dim3(256), 0, stream, node_feats, E0, E1, E2, packA, N);
  launch_gemm(packA, pw1, b1, tmpC, N, 384, 256, 1, 0, stream);
  hipLaunchKernelGGL(k_pack_a_f32, dim3(nb(N*256)), dim3(256), 0, stream, tmpC, packA, N, 256);
  launch_gemm(packA, pw2, b2, sum3, N, 256, 128, 1, 0, stream);       // sum3 as scratch
  hipLaunchKernelGGL(k_pack_a_f32, dim3(nb(N*128)), dim3(256), 0, stream, sum3, packA, N, 128);
  launch_gemm(packA, pw3, b3, h_f32, N, 128, 128, 0, 0, stream);
  hipLaunchKernelGGL(k_cast_bf16, dim3(nb(N*128)), dim3(256), 0, stream, h_f32, h_bf16, N * 128);

  // ---- SAGE layers ----
  for (int L = 0; L < NL; ++L) {
    int base = 17 + L * 19;
    hipMemsetAsync(sum3, 0, (size_t)N * 128 * 4, stream);
    const int* nbrs[2] = { nbr_fwd, nbr_bwd };
    const unsigned char* masks[2] = { mask_fwd, mask_bwd };
    for (int dir = 0; dir < 2; ++dir) {
      const float* bih = (const float*)d_in[base + dir * 7 + 2];
      const float* bhh = (const float*)d_in[base + dir * 7 + 3];
      const float* bo  = (const float*)d_in[base + dir * 7 + 6];
      hipMemsetAsync(hc, 0, (size_t)N * 128 * 4, stream);
      hipMemsetAsync(cc, 0, (size_t)N * 128 * 4, stream);
      for (int t = 0; t < D; ++t) {
        hipLaunchKernelGGL(k_pack_concat, dim3(nb(N*256)), dim3(256), 0, stream,
                           h_bf16, nbrs[dir], t, D, hc, packA, N);
        hipLaunchKernelGGL(k_lstm_step, dim3(N / 16), dim3(256), 0, stream,
                           packA, pwih[L*2+dir], bih, bhh, hc, cc, masks[dir], t, D);
      }
      // [h | hn] @ [Ws;Wn] + b  -> accumulate into sum3
      hipLaunchKernelGGL(k_pack_concat, dim3(nb(N*256)), dim3(256), 0, stream,
                         h_bf16, (const int*)nullptr, 0, D, hc, packA, N);
      launch_gemm(packA, pwsn[L*3+dir], bo, sum3, N, 256, 128, 0, 1, stream);
    }
    // mean aggregator: [h | agg] @ [Ws;Wn] + b -> accumulate
    const float* b_rep = (const float*)d_in[base + 16];
    hipLaunchKernelGGL(k_agg, dim3(nb(N*128)), dim3(256), 0, stream, h_f32, nbr_rep, mask_rep, tmpC, N, D);
    hipLaunchKernelGGL(k_pack_concat, dim3(nb(N*256)), dim3(256), 0, stream,
                       h_bf16, (const int*)nullptr, 0, D, tmpC, packA, N);
    launch_gemm(packA, pwsn[L*3+2], b_rep, sum3, N, 256, 128, 0, 1, stream);
    // h = relu(LN(sum/3 + h))
    const float* lng = (const float*)d_in[base + 17];
    const float* lnb = (const float*)d_in[base + 18];
    hipLaunchKernelGGL(k_combine_ln, dim3(N), dim3(128), 0, stream, sum3, h_f32, h_bf16, lng, lnb);
  }

  // ---- segment_max pooling + classifier ----
  hipMemsetAsync(pooled, 0, (size_t)G * 128 * 4, stream);
  hipLaunchKernelGGL(k_segmax, dim3(nb(N*128)), dim3(256), 0, stream, h_f32, graph_ids, pooled, N);
  hipLaunchKernelGGL(k_pack_a_f32, dim3(nb(G*128)), dim3(256), 0, stream, pooled, packG, G, 128);
  launch_gemm(packG, pc1, bc1, z1, G, 128, 128, 1, 0, stream);
  hipLaunchKernelGGL(k_pack_a_f32, dim3(nb(G*128)), dim3(256), 0, stream, z1, packG, G, 128);
  launch_gemm(packG, pc2, bc2, z2, G, 128, 128, 1, 0, stream);
  hipLaunchKernelGGL(k_pack_a_f32, dim3(nb(G*128)), dim3(256), 0, stream, z2, packG, G, 128);
  launch_gemm(packG, pc3, bc3, (float*)d_out, G, 128, 32, 0, 0, stream);
}
